// RNNSynthesisNet_54966991454547
// MI455X (gfx1250) — matrix operand order repaired and checked by reading, not compiled, not run on hardware
//
#include <hip/hip_runtime.h>
#include <hip/hip_bf16.h>

// ---------------------------------------------------------------------------
// Graves handwriting RNN on MI455X (gfx1250, wave32).
//  - Persistent kernel over the 600-step scan, 32 WGs, device atomic barriers.
//  - Gate GEMMs + MDN GEMM via v_wmma_f32_16x16x32_bf16 (f32 accumulate).
//  - Weights packed once to bf16 with the concat [pt|window|pad|h] folded
//    into a single K=608 (=19*32) reduction dimension.
// ---------------------------------------------------------------------------

typedef __bf16 bf16_t;
typedef __attribute__((ext_vector_type(16))) __bf16 v16bf;
typedef __attribute__((ext_vector_type(8)))  __bf16 v8bf;
typedef __attribute__((ext_vector_type(8)))  float  v8f;

#define NWG    32
#define BLOCK  256          // 8 wave32 waves
#define S_LEN  600
#define B_SZ   64
#define H_SZ   512
#define IN_SZ  3
#define C_SZ   80
#define K_MIX  20
#define U_SZ   64
#define G_SZ   2048         // 4*H
#define KPAD   608          // 3 + 80 + 13(zero pad) + 512 ; 19 * 32
#define KT_N   19           // KPAD / 32
#define NT_G   128          // G_SZ / 16
#define MT_G   4            // B_SZ / 16
#define TILES_G (MT_G * NT_G)   // 512 tiles -> 1 per wave with 512 waves... (2 per wave at 256 waves)
#define MROWS  (B_SZ * S_LEN)   // 38400
#define NPAD_MDN 128
#define OUT_RAW  121

// ---------------- workspace layout (bytes, all 256B aligned) ----------------
#define OFF_WC1   ((size_t)0)                        // 2048*608*2 = 2490368
#define OFF_WC2   ((size_t)2490368)                  // 2490368
#define OFF_WM    ((size_t)4980736)                  // 128*512*2 = 131072
#define OFF_X1    ((size_t)5111808)                  // 64*608*2  = 77824
#define OFF_X2    ((size_t)5189632)                  // 77824
#define OFF_G1    ((size_t)5267456)                  // 64*2048*4 = 524288
#define OFF_G2    ((size_t)5791744)                  // 524288
#define OFF_C1    ((size_t)6316032)                  // 64*512*4  = 131072
#define OFF_C2    ((size_t)6447104)                  // 131072
#define OFF_H1F   ((size_t)6578176)                  // 131072
#define OFF_WOUT  ((size_t)6709248)                  // 64*60*4   = 15360
#define OFF_KAPPA ((size_t)6724608)                  // 64*20*4   = 5120
#define OFF_PHI   ((size_t)6729728)                  // 64*64*4   = 16384
#define OFF_BAR   ((size_t)6746112)                  // 256 (bar @ +0, gen @ +128)
#define OFF_HS2   ((size_t)6746368)                  // 38400*512*2 = 39321600
#define OFF_MDN   ((size_t)46067968)                 // 38400*128*4 = 19660800
// total ~65.7 MB

__device__ __forceinline__ float sigf(float x) {
  return 1.0f / (1.0f + __expf(-x));
}

// ------------------------- device-wide split barrier ------------------------
__device__ __forceinline__ void grid_barrier(int* bar, int* gen) {
  __syncthreads();
  if (threadIdx.x == 0) {
    __threadfence();
    int g = __hip_atomic_load(gen, __ATOMIC_RELAXED, __HIP_MEMORY_SCOPE_AGENT);
    int prev = __hip_atomic_fetch_add(bar, 1, __ATOMIC_ACQ_REL, __HIP_MEMORY_SCOPE_AGENT);
    if (prev == (int)gridDim.x - 1) {
      __hip_atomic_store(bar, 0, __ATOMIC_RELAXED, __HIP_MEMORY_SCOPE_AGENT);
      __hip_atomic_store(gen, g + 1, __ATOMIC_RELEASE, __HIP_MEMORY_SCOPE_AGENT);
    } else {
      while (__hip_atomic_load(gen, __ATOMIC_ACQUIRE, __HIP_MEMORY_SCOPE_AGENT) == g) {
        __builtin_amdgcn_s_sleep(2);
      }
    }
    __threadfence();
  }
  __syncthreads();
}

// ------------------- bf16 WMMA gates GEMM: G = X * Wc^T ---------------------
// X: [64, KPAD] bf16 row-major, Wc: [2048, KPAD] bf16 row-major,
// G: [64, 2048] f32 row-major.
// A fragment (16-bit A 16x32): lane m=lane&15, half=lane>>4;
//   elems 0..15 = K(half*16 .. half*16+15)  (two 16B loads).
// B fragment (16-bit B 32x16): lane n=lane&15, half=lane>>4;
//   elems 0..15 = K(half*16 .. half*16+15)  (one contiguous 32B load).
__device__ __forceinline__ void gates_matmul(const bf16_t* __restrict__ X,
                                             const bf16_t* __restrict__ W,
                                             float* __restrict__ G) {
  const int lane = threadIdx.x & 31;
  const int wid  = threadIdx.x >> 5;
  const int nIdx = lane & 15;
  const int half = lane >> 4;
  const int waveId = blockIdx.x * (BLOCK / 32) + wid;
  const int wavesTotal = NWG * (BLOCK / 32);     // 256 waves -> 2 tiles each

  for (int t = waveId; t < TILES_G; t += wavesTotal) {
    const int mtile = t >> 7;       // 0..3
    const int ntile = t & (NT_G - 1);
    const bf16_t* arow = X + (size_t)(mtile * 16 + nIdx) * KPAD;
    const bf16_t* brow = W + (size_t)(ntile * 16 + nIdx) * KPAD;
    v8f acc = {};
    for (int kt = 0; kt < KT_N; ++kt) {
      __builtin_prefetch(brow + (kt + 1) * 32, 0, 1);
      v8bf a_lo = *(const v8bf*)(arow + kt * 32 + half * 16);
      v8bf a_hi = *(const v8bf*)(arow + kt * 32 + half * 16 + 8);
      v16bf a = __builtin_shufflevector(a_lo, a_hi,
                  0,1,2,3,4,5,6,7,8,9,10,11,12,13,14,15);
      v16bf b = *(const v16bf*)(brow + kt * 32 + half * 16);
      acc = __builtin_amdgcn_wmma_f32_16x16x32_bf16(
                false, a, false, b, (short)0, acc, false, false);
    }
    // D layout: vgpr r -> row (half*8 + r), col = lane&15.
    float* gout = G + (size_t)(mtile * 16 + half * 8) * G_SZ + ntile * 16 + nIdx;
#pragma unroll
    for (int r = 0; r < 8; ++r) gout[(size_t)r * G_SZ] = acc[r];
  }
}

// --------------------------- LSTM pointwise ---------------------------------
__device__ __forceinline__ void lstm_pointwise(const float* __restrict__ G,
                                               const float* __restrict__ bih,
                                               const float* __restrict__ bhh,
                                               float* __restrict__ C,
                                               float* __restrict__ h1f,   // layer1 only
                                               bf16_t* __restrict__ Xnext, // X buf receiving h (cols 96..)
                                               bf16_t* __restrict__ hs2,   // layer2 only
                                               int s, int layer1) {
  for (int e = blockIdx.x * BLOCK + threadIdx.x; e < B_SZ * H_SZ; e += NWG * BLOCK) {
    const int b  = e >> 9;
    const int hh = e & (H_SZ - 1);
    const float* g = G + (size_t)b * G_SZ;
    float gi = g[hh]              + bih[hh]              + bhh[hh];
    float gf = g[H_SZ + hh]       + bih[H_SZ + hh]       + bhh[H_SZ + hh];
    float gg = g[2 * H_SZ + hh]   + bih[2 * H_SZ + hh]   + bhh[2 * H_SZ + hh];
    float go = g[3 * H_SZ + hh]   + bih[3 * H_SZ + hh]   + bhh[3 * H_SZ + hh];
    float i = sigf(gi), f = sigf(gf), gv = tanhf(gg), o = sigf(go);
    float c = f * C[e] + i * gv;
    C[e] = c;
    float h = o * tanhf(c);
    Xnext[(size_t)b * KPAD + 96 + hh] = (bf16_t)h;
    if (layer1) {
      h1f[e] = h;
    } else {
      hs2[(size_t)(b * S_LEN + s) * H_SZ + hh] = (bf16_t)h;
    }
  }
}

// ------------------------------ params --------------------------------------
struct RnnParams {
  const float* x;       // [64,600,3]
  const float* chars;   // [64,64,80]
  const float* bih1; const float* bhh1;
  const float* Wwin; const float* bwin;
  const float* bih2; const float* bhh2;
  const bf16_t* Wc1; const bf16_t* Wc2;
  bf16_t* X1; bf16_t* X2;
  float* g1; float* g2;
  float* c1; float* c2; float* h1f;
  float* wout; float* kappa; float* phi;
  bf16_t* hs2;
  int* bar; int* gen;
};

// ----------------------- persistent recurrent kernel ------------------------
__global__ void __launch_bounds__(BLOCK)
rnn_persistent(RnnParams p) {
  const int wg  = blockIdx.x;
  const int tid = threadIdx.x;

  for (int s = 0; s < S_LEN; ++s) {
    // Phase A: gates1 = X1 * Wc1^T  (WMMA, all WGs)
    gates_matmul(p.X1, p.Wc1, p.g1);
    grid_barrier(p.bar, p.gen);

    // Phase A': LSTM1 pointwise -> h1 (f32 + bf16 into X1 h-region for t+1)
    lstm_pointwise(p.g1, p.bih1, p.bhh1, p.c1, p.h1f, p.X1, nullptr, s, 1);
    grid_barrier(p.bar, p.gen);

    // Phase B1: wout[b][j] = h1 . W_win[j] + b_win[j]   (3840 dots of 512)
    for (int e = wg * BLOCK + tid; e < B_SZ * 3 * K_MIX; e += NWG * BLOCK) {
      const int b = e / (3 * K_MIX);
      const int j = e % (3 * K_MIX);
      const float4* hv = (const float4*)(p.h1f + (size_t)b * H_SZ);
      const float4* wv = (const float4*)(p.Wwin + (size_t)j * H_SZ);
      float acc = 0.f;
      for (int q = 0; q < H_SZ / 4; ++q) {
        float4 a = hv[q], w = wv[q];
        acc += a.x * w.x + a.y * w.y + a.z * w.z + a.w * w.w;
      }
      p.wout[e] = acc + p.bwin[j];
    }
    grid_barrier(p.bar, p.gen);

    // Phase B2: Gaussian window attention (WG0 only; tiny)
    if (wg == 0) {
      // kappa += exp(k)
      for (int e = tid; e < B_SZ * K_MIX; e += BLOCK) {
        const int b = e / K_MIX, kk = e % K_MIX;
        p.kappa[e] += __expf(p.wout[b * 60 + 40 + kk]);
      }
      __syncthreads();
      // phi[b][u] = sum_k alpha * exp(-beta * (kappa-u)^2)
      for (int e = tid; e < B_SZ * U_SZ; e += BLOCK) {
        const int b = e >> 6, u = e & 63;
        const float fu = (float)u;
        float acc = 0.f;
        for (int kk = 0; kk < K_MIX; ++kk) {
          float alpha = __expf(p.wout[b * 60 + kk]);
          float beta  = __expf(p.wout[b * 60 + 20 + kk]);
          float d = p.kappa[b * K_MIX + kk] - fu;
          acc += alpha * __expf(-beta * d * d);
        }
        p.phi[e] = acc;
      }
      __syncthreads();
      // window[b][c] = sum_u phi[b][u] * chars[b][u][c]  -> bf16 into X1, X2
      for (int e = tid; e < B_SZ * C_SZ; e += BLOCK) {
        const int b = e / C_SZ, c = e % C_SZ;
        float acc = 0.f;
        for (int u = 0; u < U_SZ; ++u)
          acc += p.phi[b * U_SZ + u] * p.chars[(size_t)(b * U_SZ + u) * C_SZ + c];
        bf16_t w = (bf16_t)acc;
        p.X2[(size_t)b * KPAD + 3 + c] = w;  // LSTM2 input this step
        p.X1[(size_t)b * KPAD + 3 + c] = w;  // carried into LSTM1 next step
      }
      // pen points: current pt into X2, next pt into X1
      for (int e = tid; e < B_SZ * IN_SZ; e += BLOCK) {
        const int b = e / IN_SZ, d = e % IN_SZ;
        p.X2[(size_t)b * KPAD + d] = (bf16_t)p.x[(size_t)(b * S_LEN + s) * IN_SZ + d];
        if (s + 1 < S_LEN)
          p.X1[(size_t)b * KPAD + d] = (bf16_t)p.x[(size_t)(b * S_LEN + s + 1) * IN_SZ + d];
      }
    }
    grid_barrier(p.bar, p.gen);

    // Phase C: gates2 = X2 * Wc2^T  (WMMA, all WGs)
    gates_matmul(p.X2, p.Wc2, p.g2);
    grid_barrier(p.bar, p.gen);

    // Phase C': LSTM2 pointwise -> h2 (bf16 into X2 h-region + hs2 output)
    lstm_pointwise(p.g2, p.bih2, p.bhh2, p.c2, nullptr, p.X2, p.hs2, s, 0);
    // no trailing barrier needed: next-step phase A touches no C' outputs,
    // and barriers 1..4 of step s+1 order X2/hs2 consumers.
  }
}

// --------------------------- weight packing ---------------------------------
__global__ void pack_weights(const float* __restrict__ Wih1, const float* __restrict__ Whh1,
                             const float* __restrict__ Wih2, const float* __restrict__ Whh2,
                             const float* __restrict__ Wmdn,
                             bf16_t* __restrict__ Wc1, bf16_t* __restrict__ Wc2,
                             bf16_t* __restrict__ Wm) {
  const int tid = blockIdx.x * BLOCK + threadIdx.x;
  const int NW = G_SZ * KPAD;                    // 1245184
  if (tid < NW) {
    const int n = tid / KPAD, k = tid % KPAD;
    float v1 = 0.f, v2 = 0.f;
    if (k < 83)          { v1 = Wih1[n * 83 + k];        v2 = Wih2[n * 83 + k]; }
    else if (k >= 96)    { v1 = Whh1[n * 512 + (k - 96)]; v2 = Whh2[n * 512 + (k - 96)]; }
    Wc1[tid] = (bf16_t)v1;
    Wc2[tid] = (bf16_t)v2;
  } else {
    const int t = tid - NW;
    if (t < NPAD_MDN * H_SZ) {
      const int n = t / H_SZ, k = t % H_SZ;
      Wm[t] = (bf16_t)(n < OUT_RAW ? Wmdn[n * H_SZ + k] : 0.0f);
    }
  }
}

// ------------------------------- init ---------------------------------------
__global__ void init_state(const float* __restrict__ x,
                           bf16_t* __restrict__ X1, bf16_t* __restrict__ X2,
                           float* __restrict__ c1, float* __restrict__ c2,
                           float* __restrict__ kappa, int* bar, int* gen) {
  const int tid = blockIdx.x * BLOCK + threadIdx.x;
  if (tid < B_SZ * KPAD) {
    const int b = tid / KPAD, col = tid % KPAD;
    bf16_t v = (bf16_t)0.0f;
    if (col < IN_SZ) v = (bf16_t)x[(size_t)(b * S_LEN) * IN_SZ + col];  // pt at s=0
    X1[tid] = v;                 // window=0, pad=0, h1=0
    X2[tid] = (bf16_t)0.0f;      // fully written in-step (pt/window) ; h2=0
  }
  if (tid < B_SZ * H_SZ) { c1[tid] = 0.f; c2[tid] = 0.f; }
  if (tid < B_SZ * K_MIX) kappa[tid] = 0.f;
  if (tid == 0) { *bar = 0; *gen = 0; }
}

// ------------------------ MDN GEMM (WMMA bf16) -------------------------------
// mdnraw[38400, 128] = hs2[38400, 512] * Wm[128, 512]^T
__global__ void __launch_bounds__(BLOCK)
mdn_matmul(const bf16_t* __restrict__ hs2, const bf16_t* __restrict__ Wm,
           float* __restrict__ mdnraw) {
  const int lane = threadIdx.x & 31;
  const int wid  = threadIdx.x >> 5;
  const int mtile = blockIdx.x * (BLOCK / 32) + wid;    // 2400 m-tiles
  if (mtile >= MROWS / 16) return;
  const int nIdx = lane & 15;
  const int half = lane >> 4;

  const bf16_t* arow = hs2 + (size_t)(mtile * 16 + nIdx) * H_SZ;
  v8f acc[8];
#pragma unroll
  for (int nt = 0; nt < 8; ++nt) acc[nt] = (v8f){};

  for (int kt = 0; kt < H_SZ / 32; ++kt) {
    v8bf a_lo = *(const v8bf*)(arow + kt * 32 + half * 16);
    v8bf a_hi = *(const v8bf*)(arow + kt * 32 + half * 16 + 8);
    v16bf a = __builtin_shufflevector(a_lo, a_hi,
                0,1,2,3,4,5,6,7,8,9,10,11,12,13,14,15);
#pragma unroll
    for (int nt = 0; nt < 8; ++nt) {
      v16bf b = *(const v16bf*)(Wm + (size_t)(nt * 16 + nIdx) * H_SZ + kt * 32 + half * 16);
      acc[nt] = __builtin_amdgcn_wmma_f32_16x16x32_bf16(
                    false, a, false, b, (short)0, acc[nt], false, false);
    }
  }
#pragma unroll
  for (int nt = 0; nt < 8; ++nt) {
    float* o = mdnraw + (size_t)(mtile * 16 + half * 8) * NPAD_MDN + nt * 16 + nIdx;
#pragma unroll
    for (int r = 0; r < 8; ++r) o[(size_t)r * NPAD_MDN] = acc[nt][r];
  }
}

// ------------------------- MDN activations ----------------------------------
// d_out layout (tuple concat): eos[38400] | pi[38400*20] | mu1 | mu2 | s1 | s2 | rho
__global__ void mdn_activations(const float* __restrict__ mdnraw,
                                const float* __restrict__ bmdn,
                                float* __restrict__ out) {
  const int r = blockIdx.x * BLOCK + threadIdx.x;
  if (r >= MROWS) return;
  const float* v = mdnraw + (size_t)r * NPAD_MDN;

  out[r] = sigf(v[0] + bmdn[0]);                          // eos

  float p[K_MIX];
  float mx = -3.4e38f;
#pragma unroll
  for (int j = 0; j < K_MIX; ++j) { p[j] = v[1 + j] + bmdn[1 + j]; mx = fmaxf(mx, p[j]); }
  float sum = 0.f;
#pragma unroll
  for (int j = 0; j < K_MIX; ++j) { p[j] = __expf(p[j] - mx); sum += p[j]; }
  const float inv = 1.0f / sum;

  float* pi  = out + MROWS;
  float* mu1 = pi  + (size_t)MROWS * K_MIX;
  float* mu2 = mu1 + (size_t)MROWS * K_MIX;
  float* s1  = mu2 + (size_t)MROWS * K_MIX;
  float* s2  = s1  + (size_t)MROWS * K_MIX;
  float* rho = s2  + (size_t)MROWS * K_MIX;
#pragma unroll
  for (int j = 0; j < K_MIX; ++j) {
    pi [(size_t)r * K_MIX + j] = p[j] * inv;
    mu1[(size_t)r * K_MIX + j] = v[21 + j] + bmdn[21 + j];
    mu2[(size_t)r * K_MIX + j] = v[41 + j] + bmdn[41 + j];
    s1 [(size_t)r * K_MIX + j] = __expf(v[61 + j] + bmdn[61 + j]);
    s2 [(size_t)r * K_MIX + j] = __expf(v[81 + j] + bmdn[81 + j]);
    rho[(size_t)r * K_MIX + j] = tanhf(v[101 + j] + bmdn[101 + j]);
  }
}

// ------------------------------- launch -------------------------------------
extern "C" void kernel_launch(void* const* d_in, const int* in_sizes, int n_in,
                              void* d_out, int out_size, void* d_ws, size_t ws_size,
                              hipStream_t stream) {
  (void)in_sizes; (void)n_in; (void)out_size; (void)ws_size;
  const float* x     = (const float*)d_in[0];
  const float* chars = (const float*)d_in[1];
  const float* Wih1  = (const float*)d_in[2];
  const float* Whh1  = (const float*)d_in[3];
  const float* bih1  = (const float*)d_in[4];
  const float* bhh1  = (const float*)d_in[5];
  const float* Wwin  = (const float*)d_in[6];
  const float* bwin  = (const float*)d_in[7];
  const float* Wih2  = (const float*)d_in[8];
  const float* Whh2  = (const float*)d_in[9];
  const float* bih2  = (const float*)d_in[10];
  const float* bhh2  = (const float*)d_in[11];
  const float* Wmdn  = (const float*)d_in[12];
  const float* bmdn  = (const float*)d_in[13];

  char* ws = (char*)d_ws;
  bf16_t* Wc1   = (bf16_t*)(ws + OFF_WC1);
  bf16_t* Wc2   = (bf16_t*)(ws + OFF_WC2);
  bf16_t* Wm    = (bf16_t*)(ws + OFF_WM);
  bf16_t* X1    = (bf16_t*)(ws + OFF_X1);
  bf16_t* X2    = (bf16_t*)(ws + OFF_X2);
  float*  g1    = (float*)(ws + OFF_G1);
  float*  g2    = (float*)(ws + OFF_G2);
  float*  c1    = (float*)(ws + OFF_C1);
  float*  c2    = (float*)(ws + OFF_C2);
  float*  h1f   = (float*)(ws + OFF_H1F);
  float*  wout  = (float*)(ws + OFF_WOUT);
  float*  kappa = (float*)(ws + OFF_KAPPA);
  float*  phi   = (float*)(ws + OFF_PHI);
  int*    bar   = (int*)(ws + OFF_BAR);
  int*    gen   = (int*)(ws + OFF_BAR + 128);
  bf16_t* hs2   = (bf16_t*)(ws + OFF_HS2);
  float*  mdnraw= (float*)(ws + OFF_MDN);

  // 1) pack weights to bf16 (Wc = [W_ih | pad | W_hh], Wm padded to 128 rows)
  const int packN = G_SZ * KPAD + NPAD_MDN * H_SZ;
  pack_weights<<<(packN + BLOCK - 1) / BLOCK, BLOCK, 0, stream>>>(
      Wih1, Whh1, Wih2, Whh2, Wmdn, Wc1, Wc2, Wm);

  // 2) init recurrent state (fresh every call: deterministic)
  init_state<<<(B_SZ * KPAD + BLOCK - 1) / BLOCK, BLOCK, 0, stream>>>(
      x, X1, X2, c1, c2, kappa, bar, gen);

  // 3) persistent 600-step scan
  RnnParams p;
  p.x = x; p.chars = chars;
  p.bih1 = bih1; p.bhh1 = bhh1; p.Wwin = Wwin; p.bwin = bwin;
  p.bih2 = bih2; p.bhh2 = bhh2;
  p.Wc1 = Wc1; p.Wc2 = Wc2; p.X1 = X1; p.X2 = X2;
  p.g1 = g1; p.g2 = g2; p.c1 = c1; p.c2 = c2; p.h1f = h1f;
  p.wout = wout; p.kappa = kappa; p.phi = phi;
  p.hs2 = hs2; p.bar = bar; p.gen = gen;
  rnn_persistent<<<NWG, BLOCK, 0, stream>>>(p);

  // 4) MDN head: WMMA GEMM then activations
  mdn_matmul<<<(MROWS / 16 + (BLOCK / 32) - 1) / (BLOCK / 32), BLOCK, 0, stream>>>(
      hs2, Wm, mdnraw);
  mdn_activations<<<(MROWS + BLOCK - 1) / BLOCK, BLOCK, 0, stream>>>(
      mdnraw, bmdn, (float*)d_out);
}
